// st_NaViT_86397562126364
// MI455X (gfx1250) — compile-verified, matrix operand order checked by reading, not run on hardware
//
#include <hip/hip_runtime.h>
#include <hip/hip_bf16.h>

#define NUM_LAYERS 4
#define H_DIM 768
#define NHEAD 12
#define HDIM 64
#define E_DIM 3072
#define B_SZ 2
#define T_SZ 8
#define L_SZ 256
#define NTOK (B_SZ * T_SZ * L_SZ) /* 4096 tokens */

typedef __attribute__((ext_vector_type(16))) __bf16        v16bf;
typedef __attribute__((ext_vector_type(8)))  __bf16        v8bf;
typedef __attribute__((ext_vector_type(8)))  float         v8f;
typedef __attribute__((ext_vector_type(8)))  unsigned short v8u16;

__device__ __forceinline__ unsigned short f2bfu(float f) {
  union { float f; unsigned u; } cv; cv.f = f;
  unsigned u = cv.u;
  u += 0x7FFFu + ((u >> 16) & 1u); // round-to-nearest-even
  return (unsigned short)(u >> 16);
}
__device__ __forceinline__ __bf16 f2bf(float f) {
  unsigned short h = f2bfu(f);
  __bf16 r; __builtin_memcpy(&r, &h, 2);
  return r;
}

// bf16 WMMA fragment: per lane, elements 0-7 cover K=kb..kb+7 and elements
// 8-15 cover K=kb+16..kb+23 (kb = 8*(lane>=16))  [ISA 7.12.2].  With a
// K-contiguous row in LDS this is exactly two 16B loads (ds_load_b128).
__device__ __forceinline__ v16bf load_frag(const __bf16* p, int kb) {
  v8bf lo = *(const v8bf*)(p + kb);
  v8bf hi = *(const v8bf*)(p + kb + 16);
  return __builtin_shufflevector(lo, hi, 0, 1, 2, 3, 4, 5, 6, 7,
                                 8, 9, 10, 11, 12, 13, 14, 15);
}

// ---- CDNA5 async global->LDS copy (ASYNCcnt), via inline asm --------------
// VGLOBAL async encoding: VDST = per-lane LDS byte address, ADDR = 64-bit
// global address (GV mode).  Low 32 bits of a generic LDS pointer are the
// LDS byte offset (ISA aperture rules), so the truncation below is exact.
__device__ __forceinline__ void async_copy_b128(const void* gsrc, void* ldst) {
  unsigned lds = (unsigned)(size_t)ldst;
  asm volatile("global_load_async_to_lds_b128 %0, %1, off"
               :: "v"(lds), "v"(gsrc) : "memory");
}
__device__ __forceinline__ void wait_async0() {
  asm volatile("s_wait_asynccnt 0x0" ::: "memory");
}

// ---------------------------------------------------------------------------
// GEMM: C[M,N] = A[M,K](bf16) * B[K,N](bf16) + bias[N] + resid[M,N]
// block tile 32x128, 8 waves, wave tile 16x32 (2 WMMA per K-step of 32).
// A tile async-copied row-major [m][k]; B tile staged transposed [n][k] so
// every WMMA operand is two ds_load_b128.
// ---------------------------------------------------------------------------
__global__ __launch_bounds__(256) void gemm_bf16_wmma(
    const __bf16* __restrict__ A, const __bf16* __restrict__ B,
    const float* __restrict__ bias, const float* __restrict__ resid,
    float* __restrict__ C, int M, int N, int K)
{
  __shared__ __bf16 As[32][40];    // 80B rows (16B aligned)
  __shared__ __bf16 Bt[128][40];   // transposed [n][k]
  const int tid  = threadIdx.x;
  const int lane = tid & 31;
  const int wave = tid >> 5;
  const int wm   = wave >> 2;      // 0..1
  const int wn   = wave & 3;       // 0..3
  const int row  = lane & 15;
  const int half = lane >> 4;
  const int kb   = half * 8;
  const int blockM = blockIdx.y * 32;
  const int blockN = blockIdx.x * 128;

  v8f acc0 = {};
  v8f acc1 = {};

  for (int k0 = 0; k0 < K; k0 += 32) {
    // ---- stage A tile 32x32 via async copy (wave 0, lane -> row) ----
    if (wave == 0) {
#pragma unroll
      for (int c = 0; c < 4; ++c)  // 4 x 16B per row
        async_copy_b128(A + (size_t)(blockM + lane) * K + k0 + c * 8,
                        &As[lane][c * 8]);
    }
    // ---- stage B tile 32x128 transposed: thread owns 2k x 8n block ----
    {
      int kk = (tid & 15) * 2;
      int nn = (tid >> 4) * 8;
      v8u16 r0 = *(const v8u16*)(B + (size_t)(k0 + kk) * N + blockN + nn);
      v8u16 r1 = *(const v8u16*)(B + (size_t)(k0 + kk + 1) * N + blockN + nn);
#pragma unroll
      for (int j = 0; j < 8; ++j) {
        unsigned pk = (unsigned)r0[j] | ((unsigned)r1[j] << 16);
        *(unsigned*)&Bt[nn + j][kk] = pk;
      }
    }
    if (k0 + 32 < K) { // prefetch next K tile -> global_prefetch_b8
      __builtin_prefetch(A + (size_t)(blockM + lane) * K + k0 + 32, 0, 1);
      __builtin_prefetch(B + (size_t)(k0 + 32 + wave * 4) * N + blockN + lane * 4, 0, 1);
    }
    if (wave == 0) wait_async0();
    __syncthreads();

    v16bf af  = load_frag(&As[wm * 16 + row][0], kb);
    v16bf bf0 = load_frag(&Bt[wn * 32 + row][0], kb);
    v16bf bf1 = load_frag(&Bt[wn * 32 + 16 + row][0], kb);
    acc0 = __builtin_amdgcn_wmma_f32_16x16x32_bf16(false, af, false, bf0,
                                                   (short)0, acc0, false, false);
    acc1 = __builtin_amdgcn_wmma_f32_16x16x32_bf16(false, af, false, bf1,
                                                   (short)0, acc1, false, false);
    __syncthreads();
  }

  const int rbase = blockM + wm * 16 + half * 8;
#pragma unroll
  for (int sub = 0; sub < 2; ++sub) {
    int col = blockN + wn * 32 + sub * 16 + row;
    float bv = bias ? bias[col] : 0.f;
    v8f a = sub ? acc1 : acc0;
#pragma unroll
    for (int i = 0; i < 8; ++i) {
      size_t idx = (size_t)(rbase + i) * N + col;
      float v = a[i] + bv;
      if (resid) v += resid[idx];
      C[idx] = v;
    }
  }
}

// ---------------------------------------------------------------------------
// Spatial attention, flash-style: one (b,t,head) per blockIdx.x, 64 queries
// per block (4 waves x 16 rows), key chunks of 64.  Scores have NO 1/sqrt(d)
// scale (matches reference).  attn_mask[b,q,k] and padding_mask[b,k] applied.
// K kept [key][d] (K-contiguous for QK^T B-operand); V staged transposed
// [d][key] for the P.V B-operand; P tiles [q][key].
// ---------------------------------------------------------------------------
__global__ __launch_bounds__(128) void spatial_attn_kernel(
    const float* __restrict__ Qf,   // [NTOK,768] qk-normed
    const float* __restrict__ KVf,  // [NTOK,1536], K normed in [0,768)
    const unsigned char* __restrict__ amask,  // [B,256,256]
    const unsigned char* __restrict__ pmask,  // [B,256]
    float* __restrict__ Of)         // [NTOK,768]
{
  __shared__ __bf16 Ks[64][72];     // [key][d], 144B rows
  __shared__ __bf16 Vt[64][72];     // [d][key] transposed
  __shared__ __bf16 Ps[4][16][72];  // wave-private P tiles [q][key]

  const int g   = blockIdx.x;            // b*t*NHEAD groups
  const int h   = g % NHEAD;
  const int bt  = g / NHEAD;
  const int b   = bt / T_SZ;
  const int tokBase = bt * L_SZ;
  const int lane = threadIdx.x & 31;
  const int wave = threadIdx.x >> 5;
  const int row  = lane & 15;
  const int half = lane >> 4;
  const int kb   = half * 8;
  const int qTile = blockIdx.y * 64 + wave * 16;

  // Q fragments (A 16x64 -> two K-steps), vector loads from global
  v16bf aq0, aq1;
  {
    const float* qrow = Qf + (size_t)(tokBase + qTile + row) * H_DIM + h * HDIM;
    v8f q0 = *(const v8f*)(qrow + kb);
    v8f q1 = *(const v8f*)(qrow + kb + 16);
    v8f q2 = *(const v8f*)(qrow + 32 + kb);
    v8f q3 = *(const v8f*)(qrow + 48 + kb);
#pragma unroll
    for (int e = 0; e < 8; ++e) {
      aq0[e]     = f2bf(q0[e]);
      aq0[e + 8] = f2bf(q1[e]);
      aq1[e]     = f2bf(q2[e]);
      aq1[e + 8] = f2bf(q3[e]);
    }
  }

  float mrow[8], lrow[8];
#pragma unroll
  for (int i = 0; i < 8; ++i) { mrow[i] = -1e30f; lrow[i] = 0.f; }
  v8f o[4] = {};

  for (int chunk = 0; chunk < 4; ++chunk) {
    const int k0 = chunk * 64;
    // ---- stage K chunk [key][d] ----
    for (int idx = threadIdx.x; idx < 64 * 8; idx += 128) {
      int kr = idx >> 3, dd = (idx & 7) * 8;
      v8f f = *(const v8f*)(KVf + (size_t)(tokBase + k0 + kr) * (2 * H_DIM) +
                            h * HDIM + dd);
      v8bf o8;
#pragma unroll
      for (int j = 0; j < 8; ++j) o8[j] = f2bf(f[j]);
      *(v8bf*)&Ks[kr][dd] = o8;
    }
    // ---- stage V chunk transposed [d][key]: thread owns 2k x 8d block ----
    for (int idx = threadIdx.x; idx < 32 * 8; idx += 128) {
      int kr = (idx & 31) * 2;
      int dd = (idx >> 5) * 8;
      const float* vbase = KVf + (size_t)(tokBase + k0 + kr) * (2 * H_DIM) +
                           H_DIM + h * HDIM + dd;
      v8f f0 = *(const v8f*)vbase;
      v8f f1 = *(const v8f*)(vbase + 2 * H_DIM);
#pragma unroll
      for (int j = 0; j < 8; ++j) {
        unsigned pk = (unsigned)f2bfu(f0[j]) | ((unsigned)f2bfu(f1[j]) << 16);
        *(unsigned*)&Vt[dd + j][kr] = pk;
      }
    }
    __syncthreads();

    // S = Q @ K^T over 4 key sub-tiles of 16
    v8f s[4];
#pragma unroll
    for (int kt = 0; kt < 4; ++kt) {
      v16bf bk0 = load_frag(&Ks[kt * 16 + row][0], kb);
      v16bf bk1 = load_frag(&Ks[kt * 16 + row][32], kb);
      v8f z = {};
      z = __builtin_amdgcn_wmma_f32_16x16x32_bf16(false, aq0, false, bk0,
                                                  (short)0, z, false, false);
      z = __builtin_amdgcn_wmma_f32_16x16x32_bf16(false, aq1, false, bk1,
                                                  (short)0, z, false, false);
      s[kt] = z;
    }

    // masks
#pragma unroll
    for (int kt = 0; kt < 4; ++kt) {
      int col = k0 + kt * 16 + row;
      bool pm = pmask[b * L_SZ + col] != 0;
#pragma unroll
      for (int i = 0; i < 8; ++i) {
        int q = qTile + i + half * 8;
        bool am = amask[((size_t)b * L_SZ + q) * L_SZ + col] != 0;
        if (pm || am) s[kt][i] = -1e30f;
      }
    }

    // online softmax: row stats via shfl_xor within 16-lane row groups
#pragma unroll
    for (int i = 0; i < 8; ++i) {
      float cm = fmaxf(fmaxf(s[0][i], s[1][i]), fmaxf(s[2][i], s[3][i]));
#pragma unroll
      for (int mk = 1; mk < 16; mk <<= 1) cm = fmaxf(cm, __shfl_xor(cm, mk));
      float mn = fmaxf(mrow[i], cm);
      float sc = __expf(mrow[i] - mn);
      mrow[i] = mn;
      lrow[i] *= sc;
      o[0][i] *= sc; o[1][i] *= sc; o[2][i] *= sc; o[3][i] *= sc;
    }
#pragma unroll
    for (int kt = 0; kt < 4; ++kt) {
#pragma unroll
      for (int i = 0; i < 8; ++i) {
        float sv = s[kt][i];
        float p = (sv < -1e29f) ? 0.f : __expf(sv - mrow[i]);
        s[kt][i] = p;
        Ps[wave][i + half * 8][kt * 16 + row] = f2bf(p);
      }
    }
#pragma unroll
    for (int i = 0; i < 8; ++i) {
      float rs = s[0][i] + s[1][i] + s[2][i] + s[3][i];
#pragma unroll
      for (int mk = 1; mk < 16; mk <<= 1) rs += __shfl_xor(rs, mk);
      lrow[i] += rs;
    }

    // O += P @ V   (P: 16x64 A-matrix from wave-private LDS tile)
    v16bf ap0 = load_frag(&Ps[wave][row][0], kb);
    v16bf ap1 = load_frag(&Ps[wave][row][32], kb);
#pragma unroll
    for (int nt = 0; nt < 4; ++nt) {
      v16bf bv0 = load_frag(&Vt[nt * 16 + row][0], kb);
      v16bf bv1 = load_frag(&Vt[nt * 16 + row][32], kb);
      o[nt] = __builtin_amdgcn_wmma_f32_16x16x32_bf16(false, ap0, false, bv0,
                                                      (short)0, o[nt], false, false);
      o[nt] = __builtin_amdgcn_wmma_f32_16x16x32_bf16(false, ap1, false, bv1,
                                                      (short)0, o[nt], false, false);
    }
    __syncthreads();
  }

#pragma unroll
  for (int nt = 0; nt < 4; ++nt) {
    int c = nt * 16 + row;
#pragma unroll
    for (int i = 0; i < 8; ++i) {
      int q = qTile + i + half * 8;
      Of[(size_t)(tokBase + q) * H_DIM + h * HDIM + c] = o[nt][i] / lrow[i];
    }
  }
}

// ---------------------------------------------------------------------------
// Temporal attention: t=8, negligible FLOPs -> plain VALU. One (b,l) group
// per block; thread -> (head, query). Scale 1/sqrt(64)=0.125, no masks.
// ---------------------------------------------------------------------------
__global__ __launch_bounds__(128) void temporal_attn_kernel(
    const float* __restrict__ Qf, const float* __restrict__ KVf,
    float* __restrict__ Of)
{
  const int g = blockIdx.x;          // 0 .. B*L-1
  const int b = g >> 8, l = g & 255;
  const int w = threadIdx.x;
  if (w >= NHEAD * T_SZ) return;
  const int h = w >> 3, q = w & 7;
  const size_t tokQ = (size_t)(b * T_SZ + q) * L_SZ + l;
  const float* qv = Qf + tokQ * H_DIM + h * HDIM;

  float s[T_SZ];
  float mx = -1e30f;
  for (int k = 0; k < T_SZ; ++k) {
    size_t tokK = (size_t)(b * T_SZ + k) * L_SZ + l;
    const float* kv = KVf + tokK * (2 * H_DIM) + h * HDIM;
    float acc = 0.f;
#pragma unroll
    for (int d = 0; d < HDIM; ++d) acc += qv[d] * kv[d];
    s[k] = acc * 0.125f;
    mx = fmaxf(mx, s[k]);
  }
  float denom = 0.f;
  for (int k = 0; k < T_SZ; ++k) { s[k] = __expf(s[k] - mx); denom += s[k]; }
  const float inv = 1.f / denom;
  float* orow = Of + tokQ * H_DIM + h * HDIM;
  for (int d = 0; d < HDIM; ++d) {
    float acc = 0.f;
    for (int k = 0; k < T_SZ; ++k) {
      size_t tokK = (size_t)(b * T_SZ + k) * L_SZ + l;
      acc += s[k] * KVf[tokK * (2 * H_DIM) + H_DIM + h * HDIM + d];
    }
    orow[d] = acc * inv;
  }
}

// ---------------------------------------------------------------------------
// Per-head qk-norm: LN over 64 dims, learned per-head gamma. One warp per
// (token, head); stride allows operating on Q [.,768] or K-part of KV [.,1536]
// ---------------------------------------------------------------------------
__global__ __launch_bounds__(256) void qknorm_kernel(
    float* __restrict__ buf, int stride, const float* __restrict__ gamma)
{
  const int gw = blockIdx.x * 8 + (threadIdx.x >> 5);
  const int lane = threadIdx.x & 31;
  const int tok = gw / NHEAD;
  const int h = gw - tok * NHEAD;
  float* p = buf + (size_t)tok * stride + h * HDIM;
  float x0 = p[lane], x1 = p[lane + 32];
  float sum = x0 + x1;
#pragma unroll
  for (int mk = 16; mk; mk >>= 1) sum += __shfl_xor(sum, mk);
  float mean = sum * (1.0f / HDIM);
  float d0 = x0 - mean, d1 = x1 - mean;
  float var = d0 * d0 + d1 * d1;
#pragma unroll
  for (int mk = 16; mk; mk >>= 1) var += __shfl_xor(var, mk);
  float rstd = rsqrtf(var * (1.0f / HDIM) + 1e-5f);
  p[lane]      = d0 * rstd * gamma[h * HDIM + lane];
  p[lane + 32] = d1 * rstd * gamma[h * HDIM + lane + 32];
}

// LayerNorm over 768, no affine. One row per block, 256 threads x 3 elems.
__global__ __launch_bounds__(256) void ln768_kernel(
    const float* __restrict__ in, float* __restrict__ out)
{
  __shared__ float red[8];
  const int r = blockIdx.x, tid = threadIdx.x;
  const int lane = tid & 31, wv = tid >> 5;
  const float* p = in + (size_t)r * H_DIM;
  float v0 = p[tid], v1 = p[tid + 256], v2 = p[tid + 512];
  float sum = v0 + v1 + v2;
#pragma unroll
  for (int mk = 16; mk; mk >>= 1) sum += __shfl_xor(sum, mk);
  if (lane == 0) red[wv] = sum;
  __syncthreads();
  float tot = red[0] + red[1] + red[2] + red[3] + red[4] + red[5] + red[6] + red[7];
  float mean = tot * (1.0f / H_DIM);
  float d0 = v0 - mean, d1 = v1 - mean, d2 = v2 - mean;
  float vs = d0 * d0 + d1 * d1 + d2 * d2;
#pragma unroll
  for (int mk = 16; mk; mk >>= 1) vs += __shfl_xor(vs, mk);
  __syncthreads();
  if (lane == 0) red[wv] = vs;
  __syncthreads();
  float vtot = red[0] + red[1] + red[2] + red[3] + red[4] + red[5] + red[6] + red[7];
  float rstd = rsqrtf(vtot * (1.0f / H_DIM) + 1e-5f);
  float* q = out + (size_t)r * H_DIM;
  q[tid] = d0 * rstd; q[tid + 256] = d1 * rstd; q[tid + 512] = d2 * rstd;
}

__global__ void cast_bf16_kernel(const float* __restrict__ in,
                                 __bf16* __restrict__ out, int n)
{
  int i = blockIdx.x * 256 + threadIdx.x;
  if (i < n) out[i] = f2bf(in[i]);
}

__global__ void gelu_kernel(float* __restrict__ x, int n)
{
  int i = blockIdx.x * 256 + threadIdx.x;
  if (i < n) {
    float v = x[i];
    x[i] = 0.5f * v * (1.f + erff(v * 0.70710678118654752f)); // exact gelu
  }
}

// ---------------------------------------------------------------------------
extern "C" void kernel_launch(void* const* d_in, const int* in_sizes, int n_in,
                              void* d_out, int out_size, void* d_ws, size_t ws_size,
                              hipStream_t stream)
{
  (void)in_sizes; (void)n_in; (void)out_size; (void)ws_size;
  const float* x_in  = (const float*)d_in[0];
  const float* Wq_s  = (const float*)d_in[1];
  const float* bq_s  = (const float*)d_in[2];
  const float* Wkv_s = (const float*)d_in[3];
  const float* bkv_s = (const float*)d_in[4];
  const float* gq_s  = (const float*)d_in[5];
  const float* gk_s  = (const float*)d_in[6];
  const float* Wo_s  = (const float*)d_in[7];
  const float* Wq_t  = (const float*)d_in[8];
  const float* bq_t  = (const float*)d_in[9];
  const float* Wkv_t = (const float*)d_in[10];
  const float* bkv_t = (const float*)d_in[11];
  const float* gq_t  = (const float*)d_in[12];
  const float* gk_t  = (const float*)d_in[13];
  const float* Wo_t  = (const float*)d_in[14];
  const float* W1    = (const float*)d_in[15];
  const float* b1    = (const float*)d_in[16];
  const float* W2    = (const float*)d_in[17];
  const float* b2    = (const float*)d_in[18];
  const unsigned char* amask = (const unsigned char*)d_in[19];
  const unsigned char* pmask = (const unsigned char*)d_in[20];

  char* ws = (char*)d_ws;
  size_t off = 0;
  auto alloc = [&](size_t bytes) -> void* {
    void* p = (void*)(ws + off);
    off += (bytes + 255) & ~(size_t)255;
    return p;
  };
  float*  X0   = (float*)alloc((size_t)NTOK * H_DIM * 4);
  float*  X1   = (float*)alloc((size_t)NTOK * H_DIM * 4);
  float*  Qf   = (float*)alloc((size_t)NTOK * H_DIM * 4);
  float*  KVf  = (float*)alloc((size_t)NTOK * 2 * H_DIM * 4);
  float*  Ofb  = (float*)alloc((size_t)NTOK * H_DIM * 4);
  float*  Hf   = (float*)alloc((size_t)NTOK * E_DIM * 4);
  float*  RESf = (float*)alloc((size_t)NTOK * H_DIM * 4);
  __bf16* A16  = (__bf16*)alloc((size_t)NTOK * E_DIM * 2);
  __bf16* B16  = (__bf16*)alloc((size_t)H_DIM * E_DIM * 2);

  (void)hipMemcpyAsync(X0, x_in, (size_t)NTOK * H_DIM * 4,
                       hipMemcpyDeviceToDevice, stream);

  float* X = X0;
  float* Xn = X1;

  auto cast = [&](const float* src, __bf16* dst, int n) {
    cast_bf16_kernel<<<(n + 255) / 256, 256, 0, stream>>>(src, dst, n);
  };
  auto gemm = [&](const __bf16* A, const __bf16* B, const float* bias,
                  const float* resid, float* C, int M, int N, int K) {
    dim3 grid(N / 128, M / 32);
    gemm_bf16_wmma<<<grid, 256, 0, stream>>>(A, B, bias, resid, C, M, N, K);
  };
  auto qkn = [&](float* buf, int stride, const float* gamma) {
    qknorm_kernel<<<NTOK * NHEAD / 8, 256, 0, stream>>>(buf, stride, gamma);
  };

  for (int L = 0; L < NUM_LAYERS; ++L) {
    // -------- spatial attention block --------
    cast(X, A16, NTOK * H_DIM);
    cast(Wq_s + (size_t)L * H_DIM * H_DIM, B16, H_DIM * H_DIM);
    gemm(A16, B16, bq_s + (size_t)L * H_DIM, nullptr, Qf, NTOK, H_DIM, H_DIM);
    cast(Wkv_s + (size_t)L * H_DIM * 2 * H_DIM, B16, H_DIM * 2 * H_DIM);
    gemm(A16, B16, bkv_s + (size_t)L * 2 * H_DIM, nullptr, KVf, NTOK, 2 * H_DIM, H_DIM);
    qkn(Qf, H_DIM, gq_s + (size_t)L * NHEAD * HDIM);
    qkn(KVf, 2 * H_DIM, gk_s + (size_t)L * NHEAD * HDIM);
    {
      dim3 grid(B_SZ * T_SZ * NHEAD, 4);
      spatial_attn_kernel<<<grid, 128, 0, stream>>>(Qf, KVf, amask, pmask, Ofb);
    }
    cast(Ofb, A16, NTOK * H_DIM);
    cast(Wo_s + (size_t)L * H_DIM * H_DIM, B16, H_DIM * H_DIM);
    gemm(A16, B16, nullptr, X, Xn, NTOK, H_DIM, H_DIM);
    { float* t = X; X = Xn; Xn = t; }

    // -------- temporal attention block --------
    cast(X, A16, NTOK * H_DIM);
    cast(Wq_t + (size_t)L * H_DIM * H_DIM, B16, H_DIM * H_DIM);
    gemm(A16, B16, bq_t + (size_t)L * H_DIM, nullptr, Qf, NTOK, H_DIM, H_DIM);
    cast(Wkv_t + (size_t)L * H_DIM * 2 * H_DIM, B16, H_DIM * 2 * H_DIM);
    gemm(A16, B16, bkv_t + (size_t)L * 2 * H_DIM, nullptr, KVf, NTOK, 2 * H_DIM, H_DIM);
    qkn(Qf, H_DIM, gq_t + (size_t)L * NHEAD * HDIM);
    qkn(KVf, 2 * H_DIM, gk_t + (size_t)L * NHEAD * HDIM);
    temporal_attn_kernel<<<B_SZ * L_SZ, 128, 0, stream>>>(Qf, KVf, Ofb);
    cast(Ofb, A16, NTOK * H_DIM);
    cast(Wo_t + (size_t)L * H_DIM * H_DIM, B16, H_DIM * H_DIM);
    gemm(A16, B16, nullptr, X, Xn, NTOK, H_DIM, H_DIM);
    { float* t = X; X = Xn; Xn = t; }

    // -------- MLP block (pre-LN, no affine) --------
    ln768_kernel<<<NTOK, 256, 0, stream>>>(X, RESf);
    cast(RESf, A16, NTOK * H_DIM);
    cast(W1 + (size_t)L * H_DIM * E_DIM, B16, H_DIM * E_DIM);
    gemm(A16, B16, b1 + (size_t)L * E_DIM, nullptr, Hf, NTOK, E_DIM, H_DIM);
    gelu_kernel<<<(NTOK * E_DIM + 255) / 256, 256, 0, stream>>>(Hf, NTOK * E_DIM);
    cast(Hf, A16, NTOK * E_DIM);
    cast(W2 + (size_t)L * E_DIM * H_DIM, B16, E_DIM * H_DIM);
    gemm(A16, B16, b2 + (size_t)L * H_DIM, RESf, Ofb, NTOK, H_DIM, E_DIM);
    ln768_kernel<<<NTOK, 256, 0, stream>>>(Ofb, Xn);
    { float* t = X; X = Xn; Xn = t; }
  }

  // final input_ln applied to encoder output
  ln768_kernel<<<NTOK, 256, 0, stream>>>(X, (float*)d_out);
}
